// GNN_50757923504329
// MI455X (gfx1250) — matrix operand order repaired and checked by reading, not compiled, not run on hardware
//
#include <hip/hip_runtime.h>

#define Hd 128
#define N_NODES 50000
#define N_EDGES 600000
#define NT_NODE (N_NODES / 16)   // 3125
#define NT_EDGE (N_EDGES / 16)   // 37500

typedef __attribute__((ext_vector_type(16))) __bf16 v16bf;
typedef __attribute__((ext_vector_type(8)))  float  v8f;

// ---------------------------------------------------------------------------
// Stage a 128x128 f32 weight W[k*128+n] (k = reduction dim) into LDS as bf16,
// pre-swizzled into V_WMMA_F32_16X16X32_BF16 B-fragment order:
//   chunk(kb,nb) = 32x16 sub-matrix; within a chunk lane L owns 16 contiguous
//   bf16, where L = (n&15) + 16*((k>>3)&1), j = (k&7) + 8*((k>>4)&1).
// A wave then fetches its B operand with a single contiguous 32B LDS read.
// ---------------------------------------------------------------------------
__device__ __forceinline__ void stage_w(__bf16* dst, const float* __restrict__ W,
                                        int tid, int nthr) {
  for (int i = tid; i < Hd * Hd; i += nthr) {
    const int k = i >> 7, n = i & 127;
    const int lane = (n & 15) + (((k >> 3) & 1) << 4);
    dst[((((k >> 5) << 3) + (n >> 4)) << 9) + (lane << 4) + ((k & 7) + (((k >> 4) & 1) << 3))] = (__bf16)W[i];
  }
}

// A-fragment (16x32 bf16) for one k-block from row-major f32 [rows,128].
// ISA layout: lanes 0-15 M=0..15 K=[0..7,16..23]; lanes 16-31 M=0..15 K=+8.
__device__ __forceinline__ v16bf load_a_f32(const float* __restrict__ src, int m0,
                                            int kb, int lane) {
  const float* p = src + (m0 + (lane & 15)) * Hd + (kb << 5) + ((lane >> 4) << 3);
  v16bf a;
#pragma unroll
  for (int j = 0; j < 8; ++j) { a[j] = (__bf16)p[j]; a[j + 8] = (__bf16)p[j + 16]; }
  return a;
}

// Same, from a per-wave bf16 LDS tile [16][128].
__device__ __forceinline__ v16bf load_a_lds(const __bf16* __restrict__ t, int kb,
                                            int lane) {
  const __bf16* p = t + (lane & 15) * Hd + (kb << 5) + ((lane >> 4) << 3);
  v16bf a;
#pragma unroll
  for (int j = 0; j < 8; ++j) { a[j] = p[j]; a[j + 8] = p[j + 16]; }
  return a;
}

// One 16x16 output tile (n-block nb) over the full K=128: 4 chained WMMAs.
__device__ __forceinline__ v8f wmma_tile(const v16bf a[4], const __bf16* w, int nb,
                                         int lane) {
  v8f acc = {};
  const v16bf* wf = (const v16bf*)w;
#pragma unroll
  for (int kb = 0; kb < 4; ++kb)
    acc = __builtin_amdgcn_wmma_f32_16x16x32_bf16(
        false, a[kb], false, wf[(((kb << 3) + nb) << 5) + lane],
        (short)0, acc, false, false);
  return acc;
}

// ---------------------------------------------------------------------------
// fi = x @ Win ; fj = x @ Wout          (node projection, weights in LDS)
// ---------------------------------------------------------------------------
__global__ void __launch_bounds__(256)
proj_kernel(const float* __restrict__ x, const float* __restrict__ Win,
            const float* __restrict__ Wout, float* __restrict__ fi,
            float* __restrict__ fj, int ntiles) {
  __shared__ __align__(32) __bf16 sWa[Hd * Hd];
  __shared__ __align__(32) __bf16 sWb[Hd * Hd];
  const int tid = threadIdx.x;
  stage_w(sWa, Win, tid, blockDim.x);
  stage_w(sWb, Wout, tid, blockDim.x);
  __syncthreads();
  const int lane = tid & 31, wv = tid >> 5, nw = blockDim.x >> 5;
  const int n = lane & 15, mo = (lane >> 4) << 3;
  for (int t = blockIdx.x * nw + wv; t < ntiles; t += gridDim.x * nw) {
    const int m0 = t << 4;
    v16bf a[4];
#pragma unroll
    for (int kb = 0; kb < 4; ++kb) a[kb] = load_a_f32(x, m0, kb, lane);
#pragma unroll
    for (int nb = 0; nb < 8; ++nb) {
      v8f acc = wmma_tile(a, sWa, nb, lane);
      float* o = fi + (m0 + mo) * Hd + (nb << 4) + n;
#pragma unroll
      for (int r = 0; r < 8; ++r) o[r * Hd] = acc[r];
      acc = wmma_tile(a, sWb, nb, lane);
      o = fj + (m0 + mo) * Hd + (nb << 4) + n;
#pragma unroll
      for (int r = 0; r < 8; ++r) o[r * Hd] = acc[r];
    }
  }
}

// ---------------------------------------------------------------------------
// Fully fused edge update:
//   f  = relu(ea @ We + be + fi[row] + fj[col])
//   h  = relu(f @ W1 + b1)
//   ea' = (1+eps)*ea + h @ W2 + b2
// Three chained WMMA GEMMs per 16-edge tile; f,h live in per-wave LDS tiles.
// ---------------------------------------------------------------------------
__global__ void __launch_bounds__(256)
edge_update_kernel(const float* __restrict__ ea_in, const int* __restrict__ erow,
                   const int* __restrict__ ecol, const float* __restrict__ fi,
                   const float* __restrict__ fj, const float* __restrict__ We,
                   const float* __restrict__ be, const float* __restrict__ W1,
                   const float* __restrict__ b1, const float* __restrict__ W2,
                   const float* __restrict__ b2, const float* __restrict__ eps,
                   float* __restrict__ ea_out, int ntiles) {
  __shared__ __align__(32) __bf16 sWe[Hd * Hd];
  __shared__ __align__(32) __bf16 sW1[Hd * Hd];
  __shared__ __align__(32) __bf16 sW2[Hd * Hd];
  __shared__ __align__(32) __bf16 sF[8][16 * Hd];
  __shared__ __align__(32) __bf16 sHt[8][16 * Hd];
  const int tid = threadIdx.x;
  stage_w(sWe, We, tid, blockDim.x);
  stage_w(sW1, W1, tid, blockDim.x);
  stage_w(sW2, W2, tid, blockDim.x);
  __syncthreads();
  const int lane = tid & 31, wv = tid >> 5, nw = blockDim.x >> 5;
  __bf16* tf = sF[wv];
  __bf16* th = sHt[wv];
  const int n = lane & 15, mo = (lane >> 4) << 3;
  const float epse = 1.0f + eps[0];
  for (int t = blockIdx.x * nw + wv; t < ntiles; t += gridDim.x * nw) {
    const int e0 = t << 4;
    int ri[8], ci[8];
#pragma unroll
    for (int r = 0; r < 8; ++r) {
      ri[r] = erow[e0 + mo + r];
      ci[r] = ecol[e0 + mo + r];
    }
    v16bf a[4];
#pragma unroll
    for (int kb = 0; kb < 4; ++kb) a[kb] = load_a_f32(ea_in, e0, kb, lane);
    // f = relu(ea@We + be + gather)
#pragma unroll
    for (int nb = 0; nb < 8; ++nb) {
      v8f acc = wmma_tile(a, sWe, nb, lane);
      const int c = (nb << 4) + n;
      const float bias = be[c];
#pragma unroll
      for (int r = 0; r < 8; ++r) {
        const float v = acc[r] + bias + fi[ri[r] * Hd + c] + fj[ci[r] * Hd + c];
        tf[(mo + r) * Hd + c] = (__bf16)fmaxf(v, 0.0f);
      }
    }
    // h = relu(f@W1 + b1)
#pragma unroll
    for (int kb = 0; kb < 4; ++kb) a[kb] = load_a_lds(tf, kb, lane);
#pragma unroll
    for (int nb = 0; nb < 8; ++nb) {
      v8f acc = wmma_tile(a, sW1, nb, lane);
      const int c = (nb << 4) + n;
      const float bias = b1[c];
#pragma unroll
      for (int r = 0; r < 8; ++r)
        th[(mo + r) * Hd + c] = (__bf16)fmaxf(acc[r] + bias, 0.0f);
    }
    // ea' = (1+eps)*ea + h@W2 + b2
#pragma unroll
    for (int kb = 0; kb < 4; ++kb) a[kb] = load_a_lds(th, kb, lane);
#pragma unroll
    for (int nb = 0; nb < 8; ++nb) {
      v8f acc = wmma_tile(a, sW2, nb, lane);
      const int c = (nb << 4) + n;
      const float bias = b2[c];
#pragma unroll
      for (int r = 0; r < 8; ++r) {
        const int idx = (e0 + mo + r) * Hd + c;
        ea_out[idx] = epse * ea_in[idx] + acc[r] + bias;
      }
    }
  }
}

// ---------------------------------------------------------------------------
// msg = relu(ea @ W1 + b1) @ W2 + b2 ; agg[col[e]] += msg[e]  (atomic scatter)
// agg (25.6 MB) is L2-resident -> atomics stay on-chip.
// ---------------------------------------------------------------------------
__global__ void __launch_bounds__(256)
node_msg_kernel(const float* __restrict__ ea, const int* __restrict__ ecol,
                const float* __restrict__ W1, const float* __restrict__ b1,
                const float* __restrict__ W2, const float* __restrict__ b2,
                float* __restrict__ agg, int ntiles) {
  __shared__ __align__(32) __bf16 sW1[Hd * Hd];
  __shared__ __align__(32) __bf16 sW2[Hd * Hd];
  __shared__ __align__(32) __bf16 sT[8][16 * Hd];
  const int tid = threadIdx.x;
  stage_w(sW1, W1, tid, blockDim.x);
  stage_w(sW2, W2, tid, blockDim.x);
  __syncthreads();
  const int lane = tid & 31, wv = tid >> 5, nw = blockDim.x >> 5;
  __bf16* th = sT[wv];
  const int n = lane & 15, mo = (lane >> 4) << 3;
  for (int t = blockIdx.x * nw + wv; t < ntiles; t += gridDim.x * nw) {
    const int e0 = t << 4;
    int ci[8];
#pragma unroll
    for (int r = 0; r < 8; ++r) ci[r] = ecol[e0 + mo + r];
    v16bf a[4];
#pragma unroll
    for (int kb = 0; kb < 4; ++kb) a[kb] = load_a_f32(ea, e0, kb, lane);
#pragma unroll
    for (int nb = 0; nb < 8; ++nb) {
      v8f acc = wmma_tile(a, sW1, nb, lane);
      const int c = (nb << 4) + n;
      const float bias = b1[c];
#pragma unroll
      for (int r = 0; r < 8; ++r)
        th[(mo + r) * Hd + c] = (__bf16)fmaxf(acc[r] + bias, 0.0f);
    }
#pragma unroll
    for (int kb = 0; kb < 4; ++kb) a[kb] = load_a_lds(th, kb, lane);
#pragma unroll
    for (int nb = 0; nb < 8; ++nb) {
      v8f acc = wmma_tile(a, sW2, nb, lane);
      const int c = (nb << 4) + n;
      const float bias = b2[c];
#pragma unroll
      for (int r = 0; r < 8; ++r)
        unsafeAtomicAdd(&agg[ci[r] * Hd + c], acc[r] + bias);
    }
  }
}

// ---------------------------------------------------------------------------
// x' = (1+eps)*x + relu(agg @ W1 + b1) @ W2 + b2
// ---------------------------------------------------------------------------
__global__ void __launch_bounds__(256)
node_update_kernel(const float* __restrict__ xin, const float* __restrict__ agg,
                   const float* __restrict__ W1, const float* __restrict__ b1,
                   const float* __restrict__ W2, const float* __restrict__ b2,
                   const float* __restrict__ eps, float* __restrict__ xout,
                   int ntiles) {
  __shared__ __align__(32) __bf16 sW1[Hd * Hd];
  __shared__ __align__(32) __bf16 sW2[Hd * Hd];
  __shared__ __align__(32) __bf16 sT[8][16 * Hd];
  const int tid = threadIdx.x;
  stage_w(sW1, W1, tid, blockDim.x);
  stage_w(sW2, W2, tid, blockDim.x);
  __syncthreads();
  const int lane = tid & 31, wv = tid >> 5, nw = blockDim.x >> 5;
  __bf16* th = sT[wv];
  const int n = lane & 15, mo = (lane >> 4) << 3;
  const float epn = 1.0f + eps[0];
  for (int t = blockIdx.x * nw + wv; t < ntiles; t += gridDim.x * nw) {
    const int m0 = t << 4;
    v16bf a[4];
#pragma unroll
    for (int kb = 0; kb < 4; ++kb) a[kb] = load_a_f32(agg, m0, kb, lane);
#pragma unroll
    for (int nb = 0; nb < 8; ++nb) {
      v8f acc = wmma_tile(a, sW1, nb, lane);
      const int c = (nb << 4) + n;
      const float bias = b1[c];
#pragma unroll
      for (int r = 0; r < 8; ++r)
        th[(mo + r) * Hd + c] = (__bf16)fmaxf(acc[r] + bias, 0.0f);
    }
#pragma unroll
    for (int kb = 0; kb < 4; ++kb) a[kb] = load_a_lds(th, kb, lane);
#pragma unroll
    for (int nb = 0; nb < 8; ++nb) {
      v8f acc = wmma_tile(a, sW2, nb, lane);
      const int c = (nb << 4) + n;
      const float bias = b2[c];
#pragma unroll
      for (int r = 0; r < 8; ++r) {
        const int idx = (m0 + mo + r) * Hd + c;
        xout[idx] = epn * xin[idx] + acc[r] + bias;
      }
    }
  }
}

// ---------------------------------------------------------------------------
// Host side
// ---------------------------------------------------------------------------
extern "C" void kernel_launch(void* const* d_in, const int* in_sizes, int n_in,
                              void* d_out, int out_size, void* d_ws, size_t ws_size,
                              hipStream_t stream) {
  (void)out_size; (void)ws_size;
  const int NH = N_NODES * Hd;

  // Detect top-level flattening order (dict-insertion vs sorted-key pytree).
  const float *x0, *ea0; const int* eidx; int base;
  if (in_sizes[0] == NH) {            // insertion: x, edge_index, edge_attr, params...
    x0  = (const float*)d_in[0];
    eidx = (const int*)d_in[1];
    ea0 = (const float*)d_in[2];
    base = 3;
  } else {                            // sorted: edge_attr, edge_index, params..., x
    ea0 = (const float*)d_in[0];
    eidx = (const int*)d_in[1];
    x0  = (const float*)d_in[n_in - 1];
    base = 2;
  }
  const int* erow = eidx;
  const int* ecol = eidx + N_EDGES;

  struct LP {
    const float *We, *be, *Wi, *Wo;
    const float *eW1, *eb1, *eW2, *eb2;   // edge_mlp
    const float *aW1, *ab1, *aW2, *ab2;   // node_mlp1
    const float *uW1, *ub1, *uW2, *ub2;   // node_mlp2
    const float *eeps, *neps;
  } lp[3];
  const bool ins = (in_sizes[base + 2] == Hd * Hd);  // insertion: node_in_W; sorted: eps(1)
  for (int l = 0; l < 3; ++l) {
    const int b = base + l * 18;
    auto P = [&](int i) { return (const float*)d_in[i]; };
    if (ins) {
      lp[l] = { P(b+0),  P(b+1),  P(b+2),  P(b+3),
                P(b+4),  P(b+5),  P(b+6),  P(b+7),
                P(b+8),  P(b+9),  P(b+10), P(b+11),
                P(b+12), P(b+13), P(b+14), P(b+15),
                P(b+16), P(b+17) };
    } else {
      // sorted keys: edge_W, edge_b, edge_eps, edge_mlp{W1,W2,b1,b2},
      // node_eps, node_in_W, node_mlp1{W1,W2,b1,b2}, node_mlp2{...}, node_out_W
      lp[l].We  = P(b+0);  lp[l].be  = P(b+1);  lp[l].eeps = P(b+2);
      lp[l].eW1 = P(b+3);  lp[l].eW2 = P(b+4);  lp[l].eb1 = P(b+5);  lp[l].eb2 = P(b+6);
      lp[l].neps = P(b+7); lp[l].Wi  = P(b+8);
      lp[l].aW1 = P(b+9);  lp[l].aW2 = P(b+10); lp[l].ab1 = P(b+11); lp[l].ab2 = P(b+12);
      lp[l].uW1 = P(b+13); lp[l].uW2 = P(b+14); lp[l].ub1 = P(b+15); lp[l].ub2 = P(b+16);
      lp[l].Wo  = P(b+17);
    }
  }

  float* fi  = (float*)d_ws;          // [N,H]
  float* fj  = fi + NH;               // [N,H]
  float* agg = fj + NH;               // [N,H]
  float* xo  = (float*)d_out;         // output x  [N,H]
  float* eao = xo + NH;               // output ea [E,H]

  const float* curx  = x0;
  const float* curea = ea0;
  for (int l = 0; l < 3; ++l) {
    proj_kernel<<<dim3(256), dim3(256), 0, stream>>>(
        curx, lp[l].Wi, lp[l].Wo, fi, fj, NT_NODE);
    edge_update_kernel<<<dim3(600), dim3(256), 0, stream>>>(
        curea, erow, ecol, fi, fj, lp[l].We, lp[l].be,
        lp[l].eW1, lp[l].eb1, lp[l].eW2, lp[l].eb2, lp[l].eeps, eao, NT_EDGE);
    hipMemsetAsync(agg, 0, (size_t)NH * sizeof(float), stream);
    node_msg_kernel<<<dim3(600), dim3(256), 0, stream>>>(
        eao, ecol, lp[l].aW1, lp[l].ab1, lp[l].aW2, lp[l].ab2, agg, NT_EDGE);
    node_update_kernel<<<dim3(200), dim3(256), 0, stream>>>(
        curx, agg, lp[l].uW1, lp[l].ub1, lp[l].uW2, lp[l].ub2, lp[l].neps, xo, NT_NODE);
    curx = xo; curea = eao;
  }
}
